// _RoPESelfAttention_53334903882276
// MI455X (gfx1250) — matrix operand order repaired and checked by reading, not compiled, not run on hardware
//
#include <hip/hip_runtime.h>
#include <hip/hip_bf16.h>

// ---- types -----------------------------------------------------------------
typedef __bf16 bf16;
typedef __attribute__((ext_vector_type(16))) __bf16 v16bf;
typedef __attribute__((ext_vector_type(4)))  __bf16 v4bf;
typedef __attribute__((ext_vector_type(8)))  float  v8f;
typedef __attribute__((ext_vector_type(4)))  float  f32x4;

union AFrag { v16bf v; f32x4 q[2]; };   // 32 bytes: 16 bf16 per lane

__device__ __forceinline__ v8f wmma_bf16(const AFrag& a, const AFrag& b, v8f c) {
  return __builtin_amdgcn_wmma_f32_16x16x32_bf16(
      /*neg_a=*/false, a.v, /*neg_b=*/false, b.v,
      /*c_mod=*/(short)0, c, /*reuse_a=*/false, /*reuse_b=*/false);
}

__device__ __forceinline__ float redmax16(float v) {
  v = fmaxf(v, __shfl_xor(v, 1, 32));
  v = fmaxf(v, __shfl_xor(v, 2, 32));
  v = fmaxf(v, __shfl_xor(v, 4, 32));
  v = fmaxf(v, __shfl_xor(v, 8, 32));
  return v;
}
__device__ __forceinline__ float redsum16(float v) {
  v += __shfl_xor(v, 1, 32);
  v += __shfl_xor(v, 2, 32);
  v += __shfl_xor(v, 4, 32);
  v += __shfl_xor(v, 8, 32);
  return v;
}

// ---- constants -------------------------------------------------------------
#define BB 4
#define NN 2048
#define EE 1024
#define HH 16
#define DH 64

// ---- fp32 -> bf16 elementwise (x4 vectorized) ------------------------------
__global__ __launch_bounds__(256) void f2bf_kernel(const float* __restrict__ in,
                                                   bf16* __restrict__ out, int n) {
  int i = (blockIdx.x * 256 + threadIdx.x) * 4;
  if (i < n) {
    f32x4 f = *(const f32x4*)(in + i);
    v4bf o;
#pragma unroll
    for (int j = 0; j < 4; ++j) o[j] = (bf16)f[j];
    *(v4bf*)(out + i) = o;
  }
}

// ---- fused GEMM (+bias, +optional RoPE) ------------------------------------
// A: (8192, 1024) bf16 row-major.  W: (1024 f, 1024 e) bf16 (Linear W, e contig).
// Block: 8 waves share one 16-row m-tile (LDS-staged); each wave owns one
// 64-col head strip.  K-loop is register double-buffered.
// mode 0: bf16 (B,H,N,Dh)   (Q,K; rope=1)
// mode 1: bf16 (B,H,Dh,N)   (V transposed)
// mode 2: fp32 (m,1024)     (output projection)
__global__ __launch_bounds__(256) void proj_kernel(
    const bf16* __restrict__ A, const bf16* __restrict__ W,
    const float* __restrict__ bias, bf16* __restrict__ out_bf,
    float* __restrict__ out_f32, int mode, int rope) {
  __shared__ __align__(16) bf16 aTile[16 * EE];  // 32 KB shared x-tile

  const int lane   = threadIdx.x & 31;
  const int wid    = threadIdx.x >> 5;
  const int mtile  = blockIdx.x >> 1;                 // 512 m-tiles
  const int strip  = (blockIdx.x & 1) * 8 + wid;      // head index
  const int lane15 = lane & 15;
  const int grp    = lane >> 4;
  const int f0     = strip * DH;

  // ---- cooperative stage of the 16x1024 A tile into LDS ----
  {
    const int row   = threadIdx.x >> 4;               // 0..15
    const int chunk = threadIdx.x & 15;               // 64 halves each
    const bf16* src = A + (size_t)(mtile * 16 + row) * EE + chunk * 64;
    bf16* dst = aTile + row * EE + chunk * 64;
#pragma unroll
    for (int j = 0; j < 8; ++j)
      ((f32x4*)dst)[j] = ((const f32x4*)src)[j];
  }
  __syncthreads();

  const v8f zero = {0.f, 0.f, 0.f, 0.f, 0.f, 0.f, 0.f, 0.f};
  v8f acc[4] = {zero, zero, zero, zero};

  AFrag a[2], b[2][4];
  // preload k-step 0
  {
    const bf16* ap = &aTile[lane15 * EE + grp * 8];
    a[0].q[0] = *(const f32x4*)(ap);
    a[0].q[1] = *(const f32x4*)(ap + 16);
#pragma unroll
    for (int t = 0; t < 4; ++t) {
      const bf16* bp = W + (size_t)(f0 + t * 16 + lane15) * EE + grp * 8;
      b[0][t].q[0] = *(const f32x4*)(bp);
      b[0][t].q[1] = *(const f32x4*)(bp + 16);
    }
  }
#pragma unroll 8
  for (int ks = 0; ks < EE; ks += 32) {
    const int p  = (ks >> 5) & 1;
    const int nk = ks + 32;
    if (nk < EE) {  // prefetch next k-step into the other buffer
      const bf16* ap = &aTile[lane15 * EE + nk + grp * 8];
      a[p ^ 1].q[0] = *(const f32x4*)(ap);
      a[p ^ 1].q[1] = *(const f32x4*)(ap + 16);
#pragma unroll
      for (int t = 0; t < 4; ++t) {
        const bf16* bp = W + (size_t)(f0 + t * 16 + lane15) * EE + nk + grp * 8;
        b[p ^ 1][t].q[0] = *(const f32x4*)(bp);
        b[p ^ 1][t].q[1] = *(const f32x4*)(bp + 16);
      }
    }
#pragma unroll
    for (int t = 0; t < 4; ++t) acc[t] = wmma_bf16(a[p], b[p][t], acc[t]);
  }

  // bias (C layout: col = lane15, rows = i + grp*8)
#pragma unroll
  for (int t = 0; t < 4; ++t) {
    float bv = bias[f0 + t * 16 + lane15];
#pragma unroll
    for (int i = 0; i < 8; ++i) acc[t][i] += bv;
  }

  if (rope) {
    const float L10K_32 = 9.2103403719761836f / 32.0f;  // ln(10000)/32
    float invfA = __expf(-(float)lane15 * L10K_32);
    float invfB = __expf(-(float)(lane15 + 16) * L10K_32);
    float cs[2][8], sn[2][8];
#pragma unroll
    for (int i = 0; i < 8; ++i) {
      int m = mtile * 16 + i + grp * 8;
      float n = (float)(m & (NN - 1));
      __sincosf(n * invfA, &sn[0][i], &cs[0][i]);
      __sincosf(n * invfB, &sn[1][i], &cs[1][i]);
    }
    float r[4][8];
#pragma unroll
    for (int t = 0; t < 4; ++t) {
      const int fs = t & 1;
      const float sgn = (t < 2) ? -1.0f : 1.0f;  // rotate_half sign
#pragma unroll
      for (int i = 0; i < 8; ++i)
        r[t][i] = acc[t][i] * cs[fs][i] + sgn * acc[t ^ 2][i] * sn[fs][i];
    }
#pragma unroll
    for (int t = 0; t < 4; ++t)
#pragma unroll
      for (int i = 0; i < 8; ++i) acc[t][i] = r[t][i];
  }

#pragma unroll
  for (int t = 0; t < 4; ++t) {
#pragma unroll
    for (int i = 0; i < 8; ++i) {
      int m = mtile * 16 + i + grp * 8;
      int d = t * 16 + lane15;
      float v = acc[t][i];
      if (mode == 0) {
        int bb = m >> 11, n = m & (NN - 1);
        out_bf[(((size_t)(bb * HH + strip)) * NN + n) * DH + d] = (bf16)v;
      } else if (mode == 1) {
        int bb = m >> 11, n = m & (NN - 1);
        out_bf[(((size_t)(bb * HH + strip)) * DH + d) * NN + n] = (bf16)v;
      } else {
        out_f32[(size_t)m * EE + f0 + d] = v;
      }
    }
  }
}

// ---- flash attention: one wave = 16 queries x full Dh ----------------------
// Q,K: (B*H, N, 64) bf16.  Vt: (B*H, 64, N) bf16.  out: (B*N, 1024) bf16.
struct KF { AFrag f0a, f0b, f1a, f1b; };

__device__ __forceinline__ void loadK(KF& k, const bf16* __restrict__ Kb,
                                      int kt, int lane15, int grp) {
  const bf16* p = Kb + (size_t)(kt + lane15) * DH + grp * 8;
  k.f0a.q[0] = *(const f32x4*)(p);       k.f0a.q[1] = *(const f32x4*)(p + 16);
  k.f0b.q[0] = *(const f32x4*)(p + 32);  k.f0b.q[1] = *(const f32x4*)(p + 48);
  const bf16* p2 = p + 16 * DH;
  k.f1a.q[0] = *(const f32x4*)(p2);      k.f1a.q[1] = *(const f32x4*)(p2 + 16);
  k.f1b.q[0] = *(const f32x4*)(p2 + 32); k.f1b.q[1] = *(const f32x4*)(p2 + 48);
}

__global__ __launch_bounds__(256) void attn_kernel(
    const bf16* __restrict__ Q, const bf16* __restrict__ K,
    const bf16* __restrict__ Vt, const unsigned char* __restrict__ mask,
    bf16* __restrict__ out) {
  __shared__ __align__(16) bf16 plds[8][16 * 32];  // per-wave P staging (1 KB)

  const int lane   = threadIdx.x & 31;
  const int wid    = threadIdx.x >> 5;
  const int lane15 = lane & 15;
  const int grp    = lane >> 4;
  const int bh     = blockIdx.x >> 4;              // 64 (b,h) pairs
  const int qblk   = blockIdx.x & 15;
  const int q0     = qblk * 128 + wid * 16;
  const int b      = bh >> 4;
  const int h      = bh & 15;

  const bf16* Qb = Q  + (size_t)bh * NN * DH;
  const bf16* Kb = K  + (size_t)bh * NN * DH;
  const bf16* Vb = Vt + (size_t)bh * DH * NN;

  // Q A-fragments (d=0..31, d=32..63), held for the whole key loop
  AFrag aq0, aq1;
  {
    const bf16* p = Qb + (size_t)(q0 + lane15) * DH + grp * 8;
    aq0.q[0] = *(const f32x4*)(p);
    aq0.q[1] = *(const f32x4*)(p + 16);
    aq1.q[0] = *(const f32x4*)(p + 32);
    aq1.q[1] = *(const f32x4*)(p + 48);
  }

  float mrow[8], lrow[8];
#pragma unroll
  for (int i = 0; i < 8; ++i) { mrow[i] = -1e30f; lrow[i] = 0.f; }
  const v8f zero = {0.f, 0.f, 0.f, 0.f, 0.f, 0.f, 0.f, 0.f};
  v8f acc[4] = {zero, zero, zero, zero};

  KF kf[2];
  loadK(kf[0], Kb, 0, lane15, grp);

#pragma unroll 2
  for (int kt = 0; kt < NN; kt += 32) {
    const int p = (kt >> 5) & 1;

    // ---- S = Q K^T for 32 keys (operands already resident) ----
    v8f s0 = zero, s1 = zero;
    s0 = wmma_bf16(aq0, kf[p].f0a, s0);
    s0 = wmma_bf16(aq1, kf[p].f0b, s0);
    s1 = wmma_bf16(aq0, kf[p].f1a, s1);
    s1 = wmma_bf16(aq1, kf[p].f1b, s1);

    // ---- issue V loads + next-iteration K loads; they overlap softmax ----
    AFrag bv[4];
#pragma unroll
    for (int t = 0; t < 4; ++t) {
      const bf16* vp = Vb + (size_t)(t * 16 + lane15) * NN + kt + grp * 8;
      bv[t].q[0] = *(const f32x4*)(vp);
      bv[t].q[1] = *(const f32x4*)(vp + 16);
    }
    if (kt + 32 < NN) loadK(kf[p ^ 1], Kb, kt + 32, lane15, grp);
    if (kt + 64 < NN) {  // L2 prefetch two tiles ahead (global_prefetch_b8)
      __builtin_prefetch(Kb + (size_t)(kt + 64 + lane15) * DH, 0, 3);
      __builtin_prefetch(Vb + (size_t)lane15 * NN + kt + 64, 0, 3);
    }

    // scale + key padding mask (per C-layout column)
    const float sc = 0.125f;  // Dh^-0.5
    unsigned char mk0 = mask[b * NN + kt + lane15];
    unsigned char mk1 = mask[b * NN + kt + 16 + lane15];
#pragma unroll
    for (int i = 0; i < 8; ++i) {
      s0[i] = mk0 ? -1e30f : s0[i] * sc;
      s1[i] = mk1 ? -1e30f : s1[i] * sc;
    }

    // ---- online softmax, per row (i, lane-half) ----
#pragma unroll
    for (int i = 0; i < 8; ++i) {
      float rm   = redmax16(fmaxf(s0[i], s1[i]));
      float mnew = fmaxf(mrow[i], rm);
      float corr = __expf(mrow[i] - mnew);
      float p0 = __expf(s0[i] - mnew);
      float p1 = __expf(s1[i] - mnew);
      lrow[i] = lrow[i] * corr + redsum16(p0 + p1);
      mrow[i] = mnew;
      s0[i] = p0; s1[i] = p1;
#pragma unroll
      for (int t = 0; t < 4; ++t) acc[t][i] *= corr;
    }

    // ---- relayout P: C layout -> A layout via wave-private LDS ----
#pragma unroll
    for (int i = 0; i < 8; ++i) {
      plds[wid][(i + grp * 8) * 32 + lane15]      = (bf16)s0[i];
      plds[wid][(i + grp * 8) * 32 + 16 + lane15] = (bf16)s1[i];
    }
    AFrag pf;  // DS ops are in-order within a wave
    {
      const bf16* pp = &plds[wid][lane15 * 32 + grp * 8];
      pf.q[0] = *(const f32x4*)(pp);
      pf.q[1] = *(const f32x4*)(pp + 16);
    }

    // ---- O += P (16x32) @ V (32x64) ----
#pragma unroll
    for (int t = 0; t < 4; ++t) acc[t] = wmma_bf16(pf, bv[t], acc[t]);
  }

  // finalize + store bf16 row-major (m, e) for the output projection
#pragma unroll
  for (int t = 0; t < 4; ++t) {
#pragma unroll
    for (int i = 0; i < 8; ++i) {
      int qq = q0 + i + grp * 8;
      int d  = t * 16 + lane15;
      out[((size_t)(b * NN + qq)) * EE + h * DH + d] = (bf16)(acc[t][i] / lrow[i]);
    }
  }
}

// ---- launch ----------------------------------------------------------------
extern "C" void kernel_launch(void* const* d_in, const int* in_sizes, int n_in,
                              void* d_out, int out_size, void* d_ws, size_t ws_size,
                              hipStream_t stream) {
  const float* x  = (const float*)d_in[0];
  const float* wq = (const float*)d_in[1];
  const float* bq = (const float*)d_in[2];
  const float* wk = (const float*)d_in[3];
  const float* bk = (const float*)d_in[4];
  const float* wv = (const float*)d_in[5];
  const float* bv = (const float*)d_in[6];
  const float* wo = (const float*)d_in[7];
  const float* bo = (const float*)d_in[8];
  const unsigned char* mask = (const unsigned char*)d_in[9];
  float* out = (float*)d_out;

  char* ws = (char*)d_ws;
  const size_t MB = 1ull << 20;
  bf16* xbf  = (bf16*)(ws);                 // 16 MB : (8192,1024)
  bf16* wqbf = (bf16*)(ws + 16 * MB);       //  2 MB
  bf16* wkbf = (bf16*)(ws + 18 * MB);
  bf16* wvbf = (bf16*)(ws + 20 * MB);
  bf16* wobf = (bf16*)(ws + 22 * MB);
  bf16* Qbf  = (bf16*)(ws + 24 * MB);       // 16 MB : (B,H,N,Dh)
  bf16* Kbf  = (bf16*)(ws + 40 * MB);       // 16 MB
  bf16* Vtbf = (bf16*)(ws + 56 * MB);       // 16 MB : (B,H,Dh,N)
  bf16* AObf = (bf16*)(ws + 72 * MB);       // 16 MB : (8192,1024)

  const int nx = BB * NN * EE;              // 8388608
  const int nw = EE * EE;                   // 1048576
  f2bf_kernel<<<nx / 1024, 256, 0, stream>>>(x, xbf, nx);
  f2bf_kernel<<<nw / 1024, 256, 0, stream>>>(wq, wqbf, nw);
  f2bf_kernel<<<nw / 1024, 256, 0, stream>>>(wk, wkbf, nw);
  f2bf_kernel<<<nw / 1024, 256, 0, stream>>>(wv, wvbf, nw);
  f2bf_kernel<<<nw / 1024, 256, 0, stream>>>(wo, wobf, nw);

  // 8192 waves = 512 m-tiles x 16 heads; 8 waves (one m-tile, 8 strips) per block
  dim3 gproj(1024), blk(256);
  proj_kernel<<<gproj, blk, 0, stream>>>(xbf, wqbf, bq, Qbf,  nullptr, 0, 1);
  proj_kernel<<<gproj, blk, 0, stream>>>(xbf, wkbf, bk, Kbf,  nullptr, 0, 1);
  proj_kernel<<<gproj, blk, 0, stream>>>(xbf, wvbf, bv, Vtbf, nullptr, 1, 0);

  // 64 (b,h) x 16 q-blocks of 128 queries
  attn_kernel<<<dim3(1024), blk, 0, stream>>>(Qbf, Kbf, Vtbf, mask, AObf);

  proj_kernel<<<gproj, blk, 0, stream>>>(AObf, wobf, bo, nullptr, out, 2, 0);
}
